// MultiHeadMaskedAttention_91113436217968
// MI455X (gfx1250) — matrix operand order repaired and checked by reading, not compiled, not run on hardware
//
#include <hip/hip_runtime.h>
#include <hip/hip_bf16.h>

// ---------------- problem constants ----------------
#define BB 2
#define TT 2048
#define CC 1024
#define HH 16
#define DD 64
#define C3 (3 * CC)
#define BT (BB * TT)

typedef __attribute__((ext_vector_type(16))) __bf16 v16bf;
typedef __attribute__((ext_vector_type(8)))  float  v8f;

union FragU {
    v16bf f;
    uint4 u[2]; // 32 bytes
};

__device__ __forceinline__ v8f wmma_bf16(v16bf a, v16bf b, v8f c) {
    return __builtin_amdgcn_wmma_f32_16x16x32_bf16(
        /*neg_a=*/false, a, /*neg_b=*/false, b,
        /*c_mod=*/(short)0, c, /*reuse_a=*/false, /*reuse_b=*/false);
}

// Async 16-byte global -> LDS copy (per lane). Tracked by ASYNCcnt.
// LDS address for VDST is the low 32 bits of the generic shared pointer
// (ISA 10.2: LDS aperture maps addr[31:0] to the wave's LDS space).
__device__ __forceinline__ void async_copy16(void* lds, const void* g) {
    unsigned l = (unsigned)(uintptr_t)lds;
    unsigned long long a = (unsigned long long)(uintptr_t)g;
    asm volatile("global_load_async_to_lds_b128 %0, %1, off"
                 :: "v"(l), "v"(a) : "memory");
}
__device__ __forceinline__ void wait_async() {
    asm volatile("s_wait_asynccnt 0" ::: "memory");
}

// A-fragment: 16x32 bf16, row-major source, row stride ldx (elements).
// Lane l: row m = l&15; k-base = 8*(l>>4).  v0..3 <- k=kb..kb+7, v4..7 <- k=16+kb..16+kb+7
__device__ __forceinline__ v16bf load_a_frag(const __bf16* base, int ldx, int lane) {
    const int m  = lane & 15;
    const int kb = (lane >> 4) << 3; // 0 or 8
    const __bf16* p = base + (size_t)m * ldx + kb;
    FragU f;
    f.u[0] = *(const uint4*)(p);
    f.u[1] = *(const uint4*)(p + 16);
    return f.f;
}

// B-fragment: 32x16, element (k, n) = src[n*ldx + k]  (i.e. source is B^T row-major [N, K]).
// Lane l: col n = l&15; k-base = 8*(l>>4).  v0..3 <- k=kb..kb+7, v4..7 <- k=16+kb..16+kb+7
__device__ __forceinline__ v16bf load_bt_frag(const __bf16* base, int ldx, int lane) {
    const int n  = lane & 15;
    const int kb = (lane >> 4) << 3; // 0 or 8
    const __bf16* p = base + (size_t)n * ldx + kb;
    FragU f;
    f.u[0] = *(const uint4*)(p);
    f.u[1] = *(const uint4*)(p + 16);
    return f.f;
}

__device__ __forceinline__ __bf16 to_bf16(float x) { return (__bf16)x; }

// ---------------- prep kernels ----------------
__global__ void k_f32_to_bf16(const float* __restrict__ in, __bf16* __restrict__ out, int n) {
    int id = blockIdx.x * blockDim.x + threadIdx.x;
    if (id < n) out[id] = to_bf16(in[id]);
}

// in: [K, N] f32 row-major  ->  out: [N, K] bf16 row-major
__global__ void k_transpose_to_bf16(const float* __restrict__ in, __bf16* __restrict__ out,
                                    int K, int N) {
    int id = blockIdx.x * blockDim.x + threadIdx.x;
    if (id >= K * N) return;
    int k = id % K;
    int n = id / K;
    out[id] = to_bf16(in[(size_t)k * N + n]);
}

// qkv bf16 [B*T, 3C]  ->  vt bf16 [B, H, D, T]
__global__ void k_extract_vt(const unsigned short* __restrict__ qkv,
                             unsigned short* __restrict__ vt) {
    int id = blockIdx.x * blockDim.x + threadIdx.x; // over B*H*D*T
    int t   = id % TT;
    int rem = id / TT;
    int d   = rem % DD; rem /= DD;
    int h   = rem % HH;
    int b   = rem / HH;
    vt[id] = qkv[(size_t)(b * TT + t) * C3 + 2 * CC + h * DD + d];
}

// ---------------- WMMA GEMM:  C[M,N] = A[M,K] * B[K,N], B given transposed [N,K] ----------------
// block = 256 threads (8 waves); block tile = 128(M) x 128(N); wave tile = 16 x 128.
// B tile staged in LDS via async global->LDS DMA (double-buffered), shared by all 8 waves.
template <bool OUT_BF16>
__global__ void k_gemm_wmma(const __bf16* __restrict__ A, const __bf16* __restrict__ Bt,
                            void* __restrict__ Cout, int M, int N, int K) {
    __shared__ __align__(16) __bf16 bbuf[2][128 * 32]; // 2 x 8 KB

    const int tid  = threadIdx.x;
    const int lane = tid & 31;
    const int wave = tid >> 5;
    const int row0 = blockIdx.x * 128 + wave * 16;
    const int col0 = blockIdx.y * 128;

    v8f acc[8] = {};

    // stage Bt[col0..col0+128) x [k0..k0+32) into bbuf[p] (row-major [128][32])
    auto stage = [&](int k0, int p) {
#pragma unroll
        for (int s = 0; s < 2; ++s) {
            const int i = tid + s * 256;  // 0..511 chunks of 16B
            const int r = i >> 2;         // 0..127
            const int c = (i & 3) * 8;    // element offset 0,8,16,24
            async_copy16(&bbuf[p][r * 32 + c],
                         Bt + (size_t)(col0 + r) * K + k0 + c);
        }
    };

    stage(0, 0);
    wait_async();
    __syncthreads();

    for (int k0 = 0; k0 < K; k0 += 32) {
        const int p = (k0 >> 5) & 1;
        if (k0 + 32 < K) stage(k0 + 32, p ^ 1); // async DMA overlaps the WMMAs below

        v16bf af = load_a_frag(A + (size_t)row0 * K + k0, K, lane);
#pragma unroll
        for (int j = 0; j < 8; ++j) {
            v16bf bf = load_bt_frag(&bbuf[p][j * 16 * 32], 32, lane);
            acc[j] = wmma_bf16(af, bf, acc[j]);
        }

        wait_async();
        __syncthreads();
    }

    const int n = lane & 15, half = lane >> 4;
#pragma unroll
    for (int j = 0; j < 8; ++j) {
#pragma unroll
        for (int r = 0; r < 8; ++r) {
            const int row = row0 + r + half * 8;
            const int col = col0 + j * 16 + n;
            if (OUT_BF16)
                ((__bf16*)Cout)[(size_t)row * N + col] = to_bf16(acc[j][r]);
            else
                ((float*)Cout)[(size_t)row * N + col] = acc[j][r];
        }
    }
}

// ---------------- flash attention ----------------
// grid: B * H * (T/128) blocks; 256 threads = 8 waves; wave handles 16 query rows.
// K and V^T tiles for each 32-key block staged in LDS via async DMA (double-buffered).
__global__ void k_flash_attn(const __bf16* __restrict__ qkv, const __bf16* __restrict__ vt,
                             __bf16* __restrict__ ao) {
    __shared__ __align__(16) __bf16 kbuf[2][32 * 64]; // keys x d     : 2 x 4 KB
    __shared__ __align__(16) __bf16 vbuf[2][64 * 32]; // d x keys     : 2 x 4 KB
    __shared__ __align__(16) __bf16 pbuf[8][16 * 32]; // per-wave P   : 8 KB

    const int tid  = threadIdx.x;
    const int lane = tid & 31;
    const int wave = tid >> 5;
    int bid = blockIdx.x;
    const int qtile = bid % (TT / 128); bid /= (TT / 128);
    const int h = bid % HH;
    const int b = bid / HH;

    const int qbase = qtile * 128;
    const int q0    = qbase + wave * 16;
    const int n     = lane & 15;
    const int half  = lane >> 4;

    // Q fragments (head-dim 64 = two K=32 chunks), kept in registers
    const __bf16* qptr = qkv + (size_t)(b * TT + q0) * C3 + h * DD;
    const v16bf qf0 = load_a_frag(qptr, C3, lane);
    const v16bf qf1 = load_a_frag(qptr + 32, C3, lane);

    const __bf16* kptr = qkv + (size_t)b * TT * C3 + CC + h * DD; // K plane, row stride C3
    const __bf16* vtp  = vt + (size_t)(b * HH + h) * DD * TT;     // Vt plane [D, T]

    // stage K rows [kb..kb+32) x d[0..64) and Vt rows d[0..64) x keys [kb..kb+32)
    auto stage = [&](int kb, int p) {
        {   // kbuf: 32 rows x 128B; 8 threads of 16B per row
            const int r = tid >> 3;        // 0..31
            const int c = (tid & 7) * 8;   // 0..56
            async_copy16(&kbuf[p][r * 64 + c], kptr + (size_t)(kb + r) * C3 + c);
        }
        {   // vbuf: 64 rows x 64B; 4 threads of 16B per row
            const int r = tid >> 2;        // 0..63
            const int c = (tid & 3) * 8;   // 0..24
            async_copy16(&vbuf[p][r * 32 + c], vtp + (size_t)r * TT + kb + c);
        }
    };

    v8f o[4] = {};
    float mrow[8], lrow[8];
#pragma unroll
    for (int r = 0; r < 8; ++r) { mrow[r] = -1e30f; lrow[r] = 0.0f; }

    const float scale = 0.125f; // 1/sqrt(64)
    const int kend = qbase + 128;

    stage(0, 0);
    wait_async();
    __syncthreads();

    for (int kb = 0; kb < kend; kb += 32) {
        const int p = (kb >> 5) & 1;
        if (kb + 32 < kend) stage(kb + 32, p ^ 1); // async DMA overlaps compute below

        // ---- S = Q * K^T for 32 keys (two 16-key tiles), K-dim = 64, from LDS ----
        v8f s0 = {}, s1 = {};
        {
            v16bf b00 = load_bt_frag(&kbuf[p][0 * 64 + 0],  64, lane);
            s0 = wmma_bf16(qf0, b00, s0);
            v16bf b01 = load_bt_frag(&kbuf[p][0 * 64 + 32], 64, lane);
            s0 = wmma_bf16(qf1, b01, s0);
            v16bf b10 = load_bt_frag(&kbuf[p][16 * 64 + 0], 64, lane);
            s1 = wmma_bf16(qf0, b10, s1);
            v16bf b11 = load_bt_frag(&kbuf[p][16 * 64 + 32], 64, lane);
            s1 = wmma_bf16(qf1, b11, s1);
        }

        // ---- online softmax (row stats via cross-lane reductions in 16-lane groups) ----
        const int j0 = kb + n;       // key index of this lane, tile 0
        const int j1 = kb + 16 + n;  // tile 1
#pragma unroll
        for (int r = 0; r < 8; ++r) {
            const int qrow = q0 + r + half * 8;
            float v0 = s0[r] * scale; if (j0 > qrow) v0 = -1e30f;
            float v1 = s1[r] * scale; if (j1 > qrow) v1 = -1e30f;

            float mx = fmaxf(v0, v1);
#pragma unroll
            for (int off = 1; off < 16; off <<= 1)
                mx = fmaxf(mx, __shfl_xor(mx, off, 32));

            const float mnew  = fmaxf(mrow[r], mx);
            const float alpha = __expf(mrow[r] - mnew);
            const float e0 = __expf(v0 - mnew);
            const float e1 = __expf(v1 - mnew);

            float rs = e0 + e1;
#pragma unroll
            for (int off = 1; off < 16; off <<= 1)
                rs += __shfl_xor(rs, off, 32);

            lrow[r] = lrow[r] * alpha + rs;
            mrow[r] = mnew;
            o[0][r] *= alpha; o[1][r] *= alpha; o[2][r] *= alpha; o[3][r] *= alpha;

            // P (C-layout) -> per-wave LDS tile, row-major 16x32 bf16.
            // Wave-private buffer: same-wave DS ordering guarantees store->load.
            const int m = r + half * 8;
            pbuf[wave][m * 32 + n]      = to_bf16(e0);
            pbuf[wave][m * 32 + 16 + n] = to_bf16(e1);
        }

        // ---- O += P * V : A-frag from per-wave LDS, B-frags from staged V^T tile ----
        v16bf pf = load_a_frag(&pbuf[wave][0], 32, lane);
#pragma unroll
        for (int j = 0; j < 4; ++j) {
            v16bf bv = load_bt_frag(&vbuf[p][j * 16 * 32], 32, lane);
            o[j] = wmma_bf16(pf, bv, o[j]);
        }

        wait_async();
        __syncthreads(); // hand off: buf[p] reads done before it is restaged next iter
    }

    // ---- normalize and store attention output (bf16, [B,T,C] head-concat) ----
#pragma unroll
    for (int r = 0; r < 8; ++r) {
        const float inv = 1.0f / lrow[r];
        const int row = q0 + r + half * 8;
#pragma unroll
        for (int j = 0; j < 4; ++j) {
            ao[(size_t)(b * TT + row) * CC + h * DD + j * 16 + n] = to_bf16(o[j][r] * inv);
        }
    }
}

// ---------------- host launcher ----------------
extern "C" void kernel_launch(void* const* d_in, const int* in_sizes, int n_in,
                              void* d_out, int out_size, void* d_ws, size_t ws_size,
                              hipStream_t stream) {
    (void)in_sizes; (void)n_in; (void)out_size; (void)ws_size;

    const float* x     = (const float*)d_in[0]; // [B,T,C]
    const float* wqkv  = (const float*)d_in[1]; // [C,3C]
    const float* wout  = (const float*)d_in[2]; // [C,C]
    float*       out   = (float*)d_out;         // [B,T,C]

    // workspace carve-out (bf16 buffers)
    char* ws = (char*)d_ws;
    __bf16* xb     = (__bf16*)(ws);                       // B*T*C        =  8 MB
    __bf16* wqkvT  = (__bf16*)(ws + (8u  << 20));         // 3C*C         =  6 MB
    __bf16* woutT  = (__bf16*)(ws + (14u << 20));         // C*C          =  2 MB
    __bf16* qkv    = (__bf16*)(ws + (16u << 20));         // B*T*3C       = 24 MB
    __bf16* vt     = (__bf16*)(ws + (40u << 20));         // B*H*D*T      =  8 MB
    __bf16* ao     = (__bf16*)(ws + (48u << 20));         // B*T*C        =  8 MB

    const int nx = BT * CC;

    // 1) convert / transpose to bf16
    k_f32_to_bf16<<<nx / 256, 256, 0, stream>>>(x, xb, nx);
    k_transpose_to_bf16<<<(CC * C3) / 256, 256, 0, stream>>>(wqkv, wqkvT, CC, C3);
    k_transpose_to_bf16<<<(CC * CC) / 256, 256, 0, stream>>>(wout, woutT, CC, CC);

    // 2) qkv = x @ w_qkv  (bf16 out)
    {
        dim3 grid(BT / 128, C3 / 128);
        k_gemm_wmma<true><<<grid, 256, 0, stream>>>(xb, wqkvT, qkv, BT, C3, CC);
    }

    // 3) extract V^T  [B,H,D,T]
    k_extract_vt<<<nx / 256, 256, 0, stream>>>((const unsigned short*)qkv,
                                               (unsigned short*)vt);

    // 4) flash attention -> ao (bf16 [B,T,C])
    {
        dim3 grid(BB * HH * (TT / 128));
        k_flash_attn<<<grid, 256, 0, stream>>>(qkv, vt, ao);
    }

    // 5) out = ao @ w_out  (f32 out)
    {
        dim3 grid(BT / 128, CC / 128);
        k_gemm_wmma<false><<<grid, 256, 0, stream>>>(ao, woutT, out, BT, CC, CC);
    }
}
